// SelfAttention_44427141710372
// MI455X (gfx1250) — matrix operand order repaired and checked by reading, not compiled, not run on hardware
//
#include <hip/hip_runtime.h>
#include <hip/hip_bf16.h>
#include <cstdint>
#include <cstddef>

// Problem constants (fixed by the reference harness)
#define BB 4
#define CC 256
#define NNV 4096
#define CQ 32

typedef __attribute__((ext_vector_type(16))) __bf16 bf16x16;
typedef __attribute__((ext_vector_type(8)))  float  f32x8;
typedef __attribute__((ext_vector_type(2)))  float  f32x2;

union Frag16 {          // 16 bf16 = 32 B = 8 VGPRs, matches v16bf WMMA operand
  bf16x16 v;
  uint4   q[2];
};

__device__ __forceinline__ f32x8 wmma_bf16(bf16x16 a, bf16x16 b, f32x8 c) {
  // D = A(16x32 bf16) * B(32x16 bf16) + C(16x16 f32)
  return __builtin_amdgcn_wmma_f32_16x16x32_bf16(
      /*neg_a=*/false, a, /*neg_b=*/false, b,
      /*c_mod=*/(short)0, c, /*reuse_a=*/false, /*reuse_b=*/false);
}

// Generic pointers to LDS carry the LDS byte offset in their low 32 bits
// (ISA 10.2: LDS aperture).
__device__ __forceinline__ uint32_t lds_off_u32(const void* p) {
  return (uint32_t)(uintptr_t)p;
}

// Async DMA: global -> LDS, 16 bytes per lane, tracked by ASYNCcnt.
__device__ __forceinline__ void async_g2l_b128(uint32_t lds_byte_off,
                                               const void* gaddr) {
  asm volatile("global_load_async_to_lds_b128 %0, %1, off"
               :: "v"(lds_byte_off), "v"((unsigned long long)(uintptr_t)gaddr)
               : "memory");
}
// Async loads complete IN ORDER (ISA 08 §4.1): waiting <=4 retires the older
// of two 4-deep chunk fills while the newer chunk stays in flight.
__device__ __forceinline__ void wait_async_le4() {
  asm volatile("s_wait_asynccnt 0x4" ::: "memory");
}
__device__ __forceinline__ void wait_async_le0() {
  asm volatile("s_wait_asynccnt 0x0" ::: "memory");
}

// ---------------------------------------------------------------------------
// Kernel 0a: fp32 -> bf16 weight conversion (wq 32x256, wk 32x256, wv 256x256)
// ---------------------------------------------------------------------------
__global__ void __launch_bounds__(256) k_cvt_weights(
    const float* __restrict__ wq, const float* __restrict__ wk,
    const float* __restrict__ wv,
    __bf16* __restrict__ wqb, __bf16* __restrict__ wkb, __bf16* __restrict__ wvb) {
  int i = blockIdx.x * 256 + threadIdx.x;
  if (i < 8192)        wqb[i]         = (__bf16)wq[i];
  else if (i < 16384)  wkb[i - 8192]  = (__bf16)wk[i - 8192];
  else if (i < 81920)  wvb[i - 16384] = (__bf16)wv[i - 16384];
}

// ---------------------------------------------------------------------------
// Kernel 0b: x [B,C,N] fp32  ->  xT [B,N,C] bf16  (LDS tile transpose)
// ---------------------------------------------------------------------------
__global__ void __launch_bounds__(256) k_transpose_x(
    const float* __restrict__ x, __bf16* __restrict__ xT) {
  __shared__ float tile[32][33];
  const int b  = blockIdx.z;
  const int n0 = blockIdx.x * 32;
  const int c0 = blockIdx.y * 32;
  const float* xb = x + (size_t)b * CC * NNV;
#pragma unroll
  for (int i = threadIdx.y; i < 32; i += 8)
    tile[i][threadIdx.x] = xb[(size_t)(c0 + i) * NNV + n0 + threadIdx.x];
  __syncthreads();
  __bf16* xTb = xT + (size_t)b * NNV * CC;
#pragma unroll
  for (int i = threadIdx.y; i < 32; i += 8)
    xTb[(size_t)(n0 + i) * CC + c0 + threadIdx.x] = (__bf16)tile[threadIdx.x][i];
}

// ---------------------------------------------------------------------------
// Kernel 1: Q/K projection.  q[b,n,qc] = xT[n,:] . wq[qc,:] + bq[qc]
// One wave per 16-row n-tile; K = 256 in 8 WMMA steps; 4 accumulators.
// ---------------------------------------------------------------------------
__global__ void __launch_bounds__(256) k_proj_qk(
    const __bf16* __restrict__ xT,
    const __bf16* __restrict__ wqb, const __bf16* __restrict__ wkb,
    const float* __restrict__ bq,  const float* __restrict__ bk,
    __bf16* __restrict__ qbf, __bf16* __restrict__ kbf) {
  const int lane = threadIdx.x & 31;
  const int ln15 = lane & 15;
  const int kh   = lane >> 4;
  const int gw   = (blockIdx.x * 256 + threadIdx.x) >> 5;   // 0..1023
  const int b    = gw >> 8;                                 // 256 n-tiles / batch
  const int n0   = (gw & 255) << 4;

  const __bf16* xrow = xT + ((size_t)b * NNV + n0 + ln15) * CC;  // A rows = n

  f32x8 accq[2], acck[2];
  const f32x8 zero = {0.f,0.f,0.f,0.f,0.f,0.f,0.f,0.f};
  accq[0] = zero; accq[1] = zero; acck[0] = zero; acck[1] = zero;

  for (int ks = 0; ks < CC; ks += 32) {
    Frag16 a;
    a.q[0] = *(const uint4*)(xrow + ks + kh * 16);
    a.q[1] = *(const uint4*)(xrow + ks + kh * 16 + 8);
#pragma unroll
    for (int j = 0; j < 2; ++j) {
      Frag16 bwq, bwk;   // B[k=c][col=qc] = w[qc][c] : contiguous in c per lane
      const __bf16* wr = wqb + (size_t)(j * 16 + ln15) * CC + ks + kh * 16;
      bwq.q[0] = *(const uint4*)wr;  bwq.q[1] = *(const uint4*)(wr + 8);
      const __bf16* kr = wkb + (size_t)(j * 16 + ln15) * CC + ks + kh * 16;
      bwk.q[0] = *(const uint4*)kr;  bwk.q[1] = *(const uint4*)(kr + 8);
      accq[j] = wmma_bf16(a.v, bwq.v, accq[j]);
      acck[j] = wmma_bf16(a.v, bwk.v, acck[j]);
    }
  }
#pragma unroll
  for (int j = 0; j < 2; ++j) {
    const float bqc = bq[j * 16 + ln15];
    const float bkc = bk[j * 16 + ln15];
#pragma unroll
    for (int r = 0; r < 8; ++r) {
      const int nrow = n0 + r + 8 * kh;                    // D row = r + 8*half
      qbf[((size_t)b * NNV + nrow) * CQ + j * 16 + ln15] = (__bf16)(accq[j][r] + bqc);
      kbf[((size_t)b * NNV + nrow) * CQ + j * 16 + ln15] = (__bf16)(acck[j][r] + bkc);
    }
  }
}

// ---------------------------------------------------------------------------
// Kernel 2: V projection.  v[b,vc,n] = wv[vc,:] . xT[n,:] + bv[vc]
// One wave = 16 vc-rows x 64 n-cols (A frag reused over 4 B frags). K = 256.
// ---------------------------------------------------------------------------
__global__ void __launch_bounds__(256) k_proj_v(
    const __bf16* __restrict__ xT, const __bf16* __restrict__ wvb,
    const float* __restrict__ bv, __bf16* __restrict__ vbf) {
  const int lane = threadIdx.x & 31;
  const int ln15 = lane & 15;
  const int kh   = lane >> 4;
  const int gw   = (blockIdx.x * 256 + threadIdx.x) >> 5;   // 0..4095
  const int b    = gw >> 10;                                // 16*64 waves / batch
  const int rem  = gw & 1023;
  const int vc0  = (rem >> 6) << 4;
  const int n0   = (rem & 63) << 6;

  const __bf16* wrow = wvb + (size_t)(vc0 + ln15) * CC;     // A rows = vc

  f32x8 acc[4];
  const f32x8 zero = {0.f,0.f,0.f,0.f,0.f,0.f,0.f,0.f};
#pragma unroll
  for (int j = 0; j < 4; ++j) acc[j] = zero;

  for (int ks = 0; ks < CC; ks += 32) {
    Frag16 a;
    a.q[0] = *(const uint4*)(wrow + ks + kh * 16);
    a.q[1] = *(const uint4*)(wrow + ks + kh * 16 + 8);
#pragma unroll
    for (int j = 0; j < 4; ++j) {
      // B[k=c][col=n] = xT[n][c] : contiguous in c per lane
      const __bf16* xr =
          xT + ((size_t)b * NNV + n0 + j * 16 + ln15) * CC + ks + kh * 16;
      Frag16 bx;
      bx.q[0] = *(const uint4*)xr;  bx.q[1] = *(const uint4*)(xr + 8);
      acc[j] = wmma_bf16(a.v, bx.v, acc[j]);
    }
  }
#pragma unroll
  for (int j = 0; j < 4; ++j)
#pragma unroll
    for (int r = 0; r < 8; ++r) {
      const int c = vc0 + r + 8 * kh;
      vbf[((size_t)b * CC + c) * NNV + n0 + j * 16 + ln15] = (__bf16)(acc[j][r] + bv[c]);
    }
}

// ---------------------------------------------------------------------------
// Kernel 3: energy = q.kT (Cq=32 -> ONE wmma per 16x16 tile), row softmax,
// write attention fp32 with NT stores (268 MB stream; keep it out of L2).
// Each wave owns 16 full rows: 3 recompute sweeps (max, sum-exp, store).
// ---------------------------------------------------------------------------
__global__ void __launch_bounds__(256) k_attention(
    const __bf16* __restrict__ qbf, const __bf16* __restrict__ kbf,
    float* __restrict__ att) {
  const int lane = threadIdx.x & 31;
  const int ln15 = lane & 15;
  const int kh   = lane >> 4;
  const int gw   = (blockIdx.x * 256 + threadIdx.x) >> 5;   // 0..1023
  const int b    = gw >> 8;
  const int n0   = (gw & 255) << 4;

  Frag16 a;                                  // persistent A frag: K = Cq = 32
  const __bf16* qrow = qbf + ((size_t)b * NNV + n0 + ln15) * CQ;
  a.q[0] = *(const uint4*)(qrow + kh * 16);
  a.q[1] = *(const uint4*)(qrow + kh * 16 + 8);

  const __bf16* kb = kbf + (size_t)b * NNV * CQ;
  const f32x8 zero = {0.f,0.f,0.f,0.f,0.f,0.f,0.f,0.f};

  float mx[8], sm[8];
#pragma unroll
  for (int r = 0; r < 8; ++r) { mx[r] = -3.0e38f; sm[r] = 0.f; }

  // ---- Sweep 1: per-row max ----
  for (int m0 = 0; m0 < NNV; m0 += 16) {
    const __bf16* kr = kb + (size_t)(m0 + ln15) * CQ + kh * 16;
    Frag16 bf_;
    bf_.q[0] = *(const uint4*)kr;  bf_.q[1] = *(const uint4*)(kr + 8);
    f32x8 e = wmma_bf16(a.v, bf_.v, zero);
#pragma unroll
    for (int r = 0; r < 8; ++r) mx[r] = fmaxf(mx[r], e[r]);
  }
#pragma unroll
  for (int r = 0; r < 8; ++r)
#pragma unroll
    for (int d = 1; d < 16; d <<= 1)
      mx[r] = fmaxf(mx[r], __shfl_xor(mx[r], d, 32));

  // ---- Sweep 2: sum of exp ----
  for (int m0 = 0; m0 < NNV; m0 += 16) {
    const __bf16* kr = kb + (size_t)(m0 + ln15) * CQ + kh * 16;
    Frag16 bf_;
    bf_.q[0] = *(const uint4*)kr;  bf_.q[1] = *(const uint4*)(kr + 8);
    f32x8 e = wmma_bf16(a.v, bf_.v, zero);
#pragma unroll
    for (int r = 0; r < 8; ++r) sm[r] += __expf(e[r] - mx[r]);
  }
  float inv[8];
#pragma unroll
  for (int r = 0; r < 8; ++r) {
#pragma unroll
    for (int d = 1; d < 16; d <<= 1) sm[r] += __shfl_xor(sm[r], d, 32);
    inv[r] = __frcp_rn(sm[r]);
  }

  // ---- Sweep 3: normalize and store fp32 attention (non-temporal) ----
  float* arow = att + ((size_t)b * NNV + n0) * NNV;
  for (int m0 = 0; m0 < NNV; m0 += 16) {
    const __bf16* kr = kb + (size_t)(m0 + ln15) * CQ + kh * 16;
    Frag16 bf_;
    bf_.q[0] = *(const uint4*)kr;  bf_.q[1] = *(const uint4*)(kr + 8);
    f32x8 e = wmma_bf16(a.v, bf_.v, zero);
#pragma unroll
    for (int r = 0; r < 8; ++r)
      __builtin_nontemporal_store(__expf(e[r] - mx[r]) * inv[r],
                                  &arow[(size_t)(r + 8 * kh) * NNV + m0 + ln15]);
  }
}

// ---------------------------------------------------------------------------
// Kernel 4: out[b,c,n] = gamma * sum_m v[c,m]*att[n,m] + x[b,c,n]
// Block = 8 waves = all 256 channels x one 16-col n-tile.
//  * V chunks (256 x 32 bf16, 16 KB) DMA'd with global_load_async_to_lds_b128,
//    TRIPLE-buffered: async loads retire in order, so s_wait_asynccnt 0x4
//    retires the 2-chunks-old fill while the newer fill stays in flight ->
//    DMA latency hidden 2 chunks deep with ONE barrier per chunk.
//  * attention fp32 chunk converted to bf16 ONCE per block into LDS
//    (8x fewer v_cvt than per-wave conversion), NT loads (stream, skip L2).
//  * WMMA operands from LDS via ds_load_b128; padded row stride (40 bf16 =
//    20 banks) gives distinct starting banks for all 16 rows.
// ---------------------------------------------------------------------------
#define KC 32
#define VPITCH 40                 // 32 + 8 pad (bf16 elements)
#define NCH (NNV / KC)            // 128 chunks

__global__ void __launch_bounds__(256) k_av(
    const __bf16* __restrict__ vbf, const float* __restrict__ att,
    const float* __restrict__ x, const float* __restrict__ gamma,
    float* __restrict__ out) {
  __shared__ __bf16 sv[3][CC * VPITCH];   // V chunks   (3 x 20 KB)
  __shared__ __bf16 sb[3][16 * VPITCH];   // att chunks (3 x 1.25 KB)

  const int tid  = threadIdx.x;
  const int lane = tid & 31;
  const int ln15 = lane & 15;
  const int kh   = lane >> 4;
  const int b    = blockIdx.x >> 8;
  const int n0   = (blockIdx.x & 255) << 4;
  const int vc0  = (tid >> 5) << 5;                         // 32 channels / wave

  const __bf16* vb = vbf + (size_t)b * CC * NNV;
  const float*  ab = att + ((size_t)b * NNV + n0) * NNV;

  // DMA the 256x32 bf16 V chunk into LDS: 4 wave-wide b128 transfers per wave.
  auto fill_v_async = [&](int buf, int m0) {
#pragma unroll
    for (int j = 0; j < 4; ++j) {
      const int idx = j * 256 + tid;          // 0..1023
      const int c   = idx >> 2;
      const int ch  = (idx & 3) * 8;          // element offset within row
      const uint32_t lo = lds_off_u32(&sv[buf][c * VPITCH + ch]);
      async_g2l_b128(lo, vb + (size_t)c * NNV + m0 + ch);
    }
  };
  // Load 16x32 fp32 attention chunk (NT), convert to bf16, stage in LDS once.
  auto fill_b = [&](int buf, int m0) {
    const int row = tid >> 4;                 // 0..15
    const int cg  = (tid & 15) * 2;           // 0..30
    const f32x2 t =
        __builtin_nontemporal_load((const f32x2*)(ab + (size_t)row * NNV + m0 + cg));
    __bf16* d = &sb[buf][row * VPITCH + cg];
    d[0] = (__bf16)t.x;
    d[1] = (__bf16)t.y;
  };

  f32x8 acc0 = {0.f,0.f,0.f,0.f,0.f,0.f,0.f,0.f};
  f32x8 acc1 = acc0;

  fill_v_async(0, 0);   fill_b(0, 0);
  fill_v_async(1, KC);  fill_b(1, KC);

  int buf = 0;
  for (int ch = 0; ch < NCH; ++ch) {
    if (ch + 1 < NCH) wait_async_le4();       // oldest chunk (ch) retired
    else              wait_async_le0();       // nothing younger in flight
    __syncthreads();                          // chunk ch visible to all waves
    if (ch + 2 < NCH) {                       // refill buffer read last iteration
      const int nb = (buf + 2 >= 3) ? buf - 1 : buf + 2;
      fill_v_async(nb, (ch + 2) * KC);
      fill_b(nb, (ch + 2) * KC);
    }
    // ---- compute from LDS buffer `buf` ----
    {
      const __bf16* svb = &sv[buf][0];
      const __bf16* sbb = &sb[buf][0];
      Frag16 bf_;
      const __bf16* bp = sbb + ln15 * VPITCH + kh * 16;
      bf_.q[0] = *(const uint4*)bp;  bf_.q[1] = *(const uint4*)(bp + 8);
      Frag16 a0, a1;
      const __bf16* ap0 = svb + (vc0 + ln15) * VPITCH + kh * 16;
      a0.q[0] = *(const uint4*)ap0;  a0.q[1] = *(const uint4*)(ap0 + 8);
      const __bf16* ap1 = ap0 + 16 * VPITCH;
      a1.q[0] = *(const uint4*)ap1;  a1.q[1] = *(const uint4*)(ap1 + 8);
      acc0 = wmma_bf16(a0.v, bf_.v, acc0);
      acc1 = wmma_bf16(a1.v, bf_.v, acc1);
    }
    buf = (buf + 1 == 3) ? 0 : buf + 1;
  }

  const float gm = gamma[0];
#pragma unroll
  for (int r = 0; r < 8; ++r) {
    const int c0 = vc0 + r + 8 * kh;
    const size_t i0 = ((size_t)b * CC + c0) * NNV + n0 + ln15;
    __builtin_nontemporal_store(gm * acc0[r] + __builtin_nontemporal_load(&x[i0]),
                                &out[i0]);
    const size_t i1 = i0 + (size_t)16 * NNV;
    __builtin_nontemporal_store(gm * acc1[r] + __builtin_nontemporal_load(&x[i1]),
                                &out[i1]);
  }
}

// ---------------------------------------------------------------------------
extern "C" void kernel_launch(void* const* d_in, const int* in_sizes, int n_in,
                              void* d_out, int out_size, void* d_ws, size_t ws_size,
                              hipStream_t stream) {
  (void)in_sizes; (void)n_in; (void)out_size; (void)ws_size;
  const float* x     = (const float*)d_in[0];
  const float* wq    = (const float*)d_in[1];
  const float* bq    = (const float*)d_in[2];
  const float* wk    = (const float*)d_in[3];
  const float* bk    = (const float*)d_in[4];
  const float* wv    = (const float*)d_in[5];
  const float* bv    = (const float*)d_in[6];
  const float* gamma = (const float*)d_in[7];

  float* out = (float*)d_out;                                   // [B,C,H,W]
  float* att = out + (size_t)BB * CC * NNV;                     // [B,N,N]

  // Workspace carve (~19 MB)
  char* ws = (char*)d_ws;
  __bf16* xT  = (__bf16*)ws;                                    // [B,N,C]   8 MB
  __bf16* wqb = (__bf16*)(ws + (size_t)BB * NNV * CC * 2);      // [32,256]
  __bf16* wkb = wqb + 32 * CC;
  __bf16* wvb = wkb + 32 * CC;                                  // [256,256]
  __bf16* qbf = wvb + CC * CC;                                  // [B,N,32]  1 MB
  __bf16* kbf = qbf + (size_t)BB * NNV * CQ;                    // [B,N,32]  1 MB
  __bf16* vbf = kbf + (size_t)BB * NNV * CQ;                    // [B,C,N]   8 MB

  k_cvt_weights<<<320, 256, 0, stream>>>(wq, wk, wv, wqb, wkb, wvb);
  k_transpose_x<<<dim3(NNV / 32, CC / 32, BB), dim3(32, 8), 0, stream>>>(x, xT);
  k_proj_qk<<<128, 256, 0, stream>>>(xT, wqb, wkb, bq, bk, qbf, kbf);
  k_proj_v<<<512, 256, 0, stream>>>(xT, wvb, bv, vbf);
  k_attention<<<128, 256, 0, stream>>>(qbf, kbf, att);
  k_av<<<BB * (NNV / 16), 256, 0, stream>>>(vbf, att, x, gamma, out);
}
